// MaskedCausalAttention_91182155694536
// MI455X (gfx1250) — matrix operand order repaired and checked by reading, not compiled, not run on hardware
//
#include <hip/hip_runtime.h>

#define BATCH 4
#define SEQ   2048
#define CDIM  1024
#define NHEAD 16
#define HD    64
#define MROWS (BATCH * SEQ) // 8192

typedef __attribute__((ext_vector_type(16))) __bf16 v16bf;
typedef __attribute__((ext_vector_type(8)))  float  v8f;
typedef __attribute__((ext_vector_type(4)))  unsigned int u32x4;
typedef __attribute__((ext_vector_type(8)))  unsigned int u32x8;

union ABFrag { v16bf v; uint4 q[2]; };

__device__ inline unsigned short f2bf(float f) {
  unsigned int u = __float_as_uint(f);
  u += 0x7FFFu + ((u >> 16) & 1u);   // round-to-nearest-even
  return (unsigned short)(u >> 16);
}

__device__ inline v8f wmma_bf16(const ABFrag& a, const ABFrag& b, v8f c) {
  return __builtin_amdgcn_wmma_f32_16x16x32_bf16(
      false, a.v, false, b.v, (short)0, c, false, false);
}

// ---- TDM: 2D tile load global->LDS, 2-byte elems, LDS rows padded +16B ----
// pad_interval code 3 = 16 dwords (64B = 32 elems) ; pad_amount code 3 = 4 dwords
// => LDS row stride = 32 + 8 = 40 elems (LDAB)
// NOTE: the LDS destination pointer is passed through the descriptor (ptrtoint)
// so the shared allocation escapes into the asm -> the "memory" clobber keeps
// the subsequent ds_load fragment reads alive, and the low 32 bits of the
// LDS-aperture generic address are the real LDS byte offset.
__device__ inline void tdm_tile(unsigned short* ldsPtr,
                                const unsigned short* gptr,
                                unsigned tensor_d0, unsigned tensor_d1,
                                unsigned tile_d0, unsigned tile_d1) {
  unsigned long long ga = (unsigned long long)(const void*)gptr;
  unsigned lds_byte = (unsigned)(unsigned long long)(void*)ldsPtr;
  u32x4 g0;
  g0[0] = 1u;                                    // count = 1 (valid user D#)
  g0[1] = lds_byte;                              // lds_addr
  g0[2] = (unsigned)ga;                          // global_addr[31:0]
  g0[3] = (unsigned)((ga >> 32) & 0x01FFFFFFu)   // global_addr[56:32]
        | (2u << 30);                            // type = 2 ("image")
  u32x8 g1;
  g1[0] = (1u << 16)                             // data_size = 2 bytes
        | (1u << 20)                             // pad_enable
        | (3u << 22)                             // pad_interval: 16 dwords
        | (3u << 25);                            // pad_amount: 4 dwords
  g1[1] = (tensor_d0 & 0xFFFFu) << 16;           // tensor_dim0[15:0]
  g1[2] = (tensor_d0 >> 16) | ((tensor_d1 & 0xFFFFu) << 16);
  g1[3] = (tensor_d1 >> 16) | ((tile_d0 & 0xFFFFu) << 16);
  g1[4] = tile_d1 & 0xFFFFu;                     // tile_dim1 (tile_dim2 = 0)
  g1[5] = tensor_d0;                             // tensor_dim0_stride[31:0]
  g1[6] = 0u;
  g1[7] = 0u;
  asm volatile("tensor_load_to_lds %0, %1" :: "s"(g0), "s"(g1) : "memory");
}

// ---------------- fp32 -> bf16 conversion ----------------
__global__ void cvt_kernel(const float* __restrict__ in,
                           unsigned short* __restrict__ out, int n4) {
  int i = blockIdx.x * blockDim.x + threadIdx.x;
  int stride = gridDim.x * blockDim.x;
  for (; i < n4; i += stride) {
    float4 f = ((const float4*)in)[i];
    ushort4 o;
    o.x = f2bf(f.x); o.y = f2bf(f.y); o.z = f2bf(f.z); o.w = f2bf(f.w);
    ((ushort4*)out)[i] = o;
  }
}

// ---------------- GEMM: Y = X @ W^T (+bias), TDM double-buffered ----------------
// X: [MROWS, CDIM] bf16, W: [CDIM, CDIM] bf16 (row n = output col n)
// mode 0..2: scatter bf16 to [B, NHEAD, T, HD]; mode 3: fp32 [MROWS, CDIM]
#define LDAB 40
#define ABUF (128 * LDAB)
#define BBUF (64 * LDAB)
#define BUFE (ABUF + BBUF)
__global__ __launch_bounds__(256) void gemm_xwt(
    const unsigned short* __restrict__ Xb,
    const unsigned short* __restrict__ Wb,
    const float* __restrict__ bias,
    unsigned short* __restrict__ dstBf,
    float* __restrict__ dstF,
    int mode, float outScale)
{
  __shared__ unsigned short sAB[2 * BUFE];   // double-buffered A(128x32)+B(64x32)
  const int tid  = threadIdx.x;
  const int lane = tid & 31;
  const int wv   = tid >> 5;                  // 8 waves
  const int m0 = blockIdx.y * 128;
  const int n0 = blockIdx.x * 64;

  v8f zero8 = {0.f,0.f,0.f,0.f,0.f,0.f,0.f,0.f};
  v8f acc[4];
  for (int n = 0; n < 4; ++n) acc[n] = zero8;

  // prime the TDM pipeline: tiles for k0 = 0 into buffer 0
  if (wv == 0) {
    tdm_tile(sAB,        Xb + (size_t)m0 * CDIM, CDIM, MROWS, 32, 128);
    tdm_tile(sAB + ABUF, Wb + (size_t)n0 * CDIM, CDIM, CDIM,  32, 64);
  }

  int cur = 0;
  for (int k0 = 0; k0 < CDIM; k0 += 32) {
    if (wv == 0) __builtin_amdgcn_s_wait_tensorcnt(0);
    __syncthreads();   // tiles[cur] ready; all waves done reading tiles[cur^1]

    if (wv == 0 && k0 + 32 < CDIM) {         // async prefetch next K-tiles
      unsigned short* nb = sAB + (cur ^ 1) * BUFE;
      tdm_tile(nb,        Xb + (size_t)m0 * CDIM + k0 + 32, CDIM, MROWS, 32, 128);
      tdm_tile(nb + ABUF, Wb + (size_t)n0 * CDIM + k0 + 32, CDIM, CDIM,  32, 64);
    }

    const unsigned short* cA = sAB + cur * BUFE;
    const unsigned short* cB = cA + ABUF;

    // A fragment (16x32): lane<16 holds K 0..7 & 16..23, lane>=16 K 8..15 & 24..31
    ABFrag afr;
    const int arow = wv * 16 + (lane & 15);
    const int kh = (lane >> 4) * 8;
    afr.q[0] = *(const uint4*)&cA[arow * LDAB + kh];
    afr.q[1] = *(const uint4*)&cA[arow * LDAB + kh + 16];
#pragma unroll
    for (int n = 0; n < 4; ++n) {
      // B fragment (32x16): lane holds column (lane&15), lane<16 K 0..7 & 16..23, lane>=16 K 8..15 & 24..31
      ABFrag bfr;
      const int bcol = n * 16 + (lane & 15);
      const int bko  = (lane >> 4) * 8;
      bfr.q[0] = *(const uint4*)&cB[bcol * LDAB + bko];
      bfr.q[1] = *(const uint4*)&cB[bcol * LDAB + bko + 16];
      acc[n] = wmma_bf16(afr, bfr, acc[n]);
    }
    cur ^= 1;
  }

#pragma unroll
  for (int n = 0; n < 4; ++n) {
    const int gcol = n0 + n * 16 + (lane & 15);
    const float bb = bias[gcol];
#pragma unroll
    for (int r = 0; r < 8; ++r) {
      const int grow = m0 + wv * 16 + r + ((lane >> 4) << 3);
      const float y = (acc[n][r] + bb) * outScale;
      if (mode < 3) {
        const int b = grow >> 11, t = grow & (SEQ - 1);
        const int head = gcol >> 6, d = gcol & (HD - 1);
        dstBf[((size_t)(b * NHEAD + head) * SEQ + t) * HD + d] = f2bf(y);
      } else {
        dstF[(size_t)grow * CDIM + gcol] = y;
      }
    }
  }
}

// ---------------- flash attention ----------------
#define LDK 72
#define LDV 40
#define LDP 40
__global__ __launch_bounds__(128) void attn_kernel(
    const unsigned short* __restrict__ Qb,   // [B,N,T,D] bf16, pre-scaled by 1/sqrt(D)
    const unsigned short* __restrict__ Kb,
    const unsigned short* __restrict__ Vb,
    unsigned short* __restrict__ attnb)      // [B,T,C] bf16
{
  __shared__ unsigned short smem[32 * LDK + 64 * LDV + 64 * LDP];
  unsigned short* sK = smem;                     // [key][d]
  unsigned short* sV = smem + 32 * LDK;          // transposed: [d][key]
  unsigned short* sP = smem + 32 * LDK + 64 * LDV; // per-wave 16x32 P tiles
  const int tid  = threadIdx.x;
  const int lane = tid & 31;
  const int wv   = tid >> 5;
  const int q0 = blockIdx.x * 64;
  const int bn = blockIdx.y;                 // b*NHEAD + h
  const size_t headBase = (size_t)bn * SEQ * HD;

  const int kh = (lane >> 4) * 8;
  const int qrow = q0 + wv * 16 + (lane & 15);
  ABFrag qa[2];
#pragma unroll
  for (int f = 0; f < 2; ++f) {
    const unsigned short* g = Qb + headBase + (size_t)qrow * HD + f * 32 + kh;
    qa[f].q[0] = *(const uint4*)g;
    qa[f].q[1] = *(const uint4*)(g + 16);
  }

  v8f zero8 = {0.f,0.f,0.f,0.f,0.f,0.f,0.f,0.f};
  v8f acc[4];
  float mrow[8], lrow[8];
  int gr[8];
#pragma unroll
  for (int n = 0; n < 4; ++n) acc[n] = zero8;
#pragma unroll
  for (int r = 0; r < 8; ++r) {
    mrow[r] = -3.0e38f; lrow[r] = 0.f;
    gr[r] = q0 + wv * 16 + r + ((lane >> 4) << 3);
  }

  for (int s0 = 0; s0 < q0 + 64; s0 += 32) {
    __syncthreads();
    {   // K tile via per-lane async global->LDS; V loaded + transposed manually
      const int row = tid >> 2;
      const int seg = (tid & 3) * 16;
      // per-lane LDS address from the real pointer (captures smem for aliasing,
      // low 32 bits of the LDS-aperture address = LDS byte offset)
      const unsigned ldsK =
          (unsigned)(unsigned long long)(void*)(sK + row * LDK + seg);
      const unsigned long long gk =
          (unsigned long long)(Kb + headBase + (size_t)(s0 + row) * HD + seg);
      asm volatile("global_load_async_to_lds_b128 %0, %1, off"
                   :: "v"(ldsK), "v"(gk) : "memory");
      asm volatile("global_load_async_to_lds_b128 %0, %1, off offset:16"
                   :: "v"(ldsK + 16u), "v"(gk) : "memory");

      const unsigned short* gv = Vb + headBase + (size_t)(s0 + row) * HD + seg;
      __align__(16) unsigned short tv[16];
      *(uint4*)tv       = *(const uint4*)gv;
      *(uint4*)(tv + 8) = *(const uint4*)(gv + 8);
#pragma unroll
      for (int j = 0; j < 16; ++j)
        sV[(seg + j) * LDV + row] = tv[j];

      asm volatile("s_wait_asynccnt 0x0" ::: "memory");
    }
    __syncthreads();

    // S = Q @ K^T for two 16-key subtiles (1/sqrt(D) folded into Q)
    v8f S[2];
#pragma unroll
    for (int u = 0; u < 2; ++u) {
      v8f s = zero8;
#pragma unroll
      for (int f = 0; f < 2; ++f) {
        ABFrag bfr;
        const int key = u * 16 + (lane & 15);
        const int ko  = f * 32 + ((lane >> 4) * 8);
        bfr.q[0] = *(const uint4*)&sK[key * LDK + ko];
        bfr.q[1] = *(const uint4*)&sK[key * LDK + ko + 16];
        s = wmma_bf16(qa[f], bfr, s);
      }
      S[u] = s;
    }

    // causal mask + online softmax (rows live in 16-lane halves)
    float alpha[8];
    const int c0 = s0 + (lane & 15);
#pragma unroll
    for (int r = 0; r < 8; ++r) {
      float s0v = (c0      <= gr[r]) ? S[0][r] : -3.0e38f;
      float s1v = (c0 + 16 <= gr[r]) ? S[1][r] : -3.0e38f;
      float mx = fmaxf(s0v, s1v);
#pragma unroll
      for (int off = 8; off >= 1; off >>= 1)
        mx = fmaxf(mx, __shfl_xor(mx, off, 32));
      const float mnew = fmaxf(mrow[r], mx);
      const float a  = __expf(mrow[r] - mnew);
      const float p0 = __expf(s0v - mnew);
      const float p1 = __expf(s1v - mnew);
      float sum = p0 + p1;
#pragma unroll
      for (int off = 8; off >= 1; off >>= 1)
        sum += __shfl_xor(sum, off, 32);
      lrow[r] = lrow[r] * a + sum;
      mrow[r] = mnew;
      alpha[r] = a;
      const int prow = wv * 16 + r + ((lane >> 4) << 3);
      sP[prow * LDP + (lane & 15)]      = f2bf(p0);   // C-layout -> LDS
      sP[prow * LDP + 16 + (lane & 15)] = f2bf(p1);
    }
#pragma unroll
    for (int n = 0; n < 4; ++n)
#pragma unroll
      for (int r = 0; r < 8; ++r)
        acc[n][r] *= alpha[r];

    // wave-lockstep re-layout barrier: all 32 lanes' LDS stores must land
    asm volatile("s_wait_dscnt 0x0" ::: "memory");

    // acc += P @ V  (P reloaded in A-fragment layout, V transposed in LDS)
    ABFrag pa;
    const int pr = wv * 16 + (lane & 15);
    pa.q[0] = *(const uint4*)&sP[pr * LDP + kh];
    pa.q[1] = *(const uint4*)&sP[pr * LDP + kh + 16];
#pragma unroll
    for (int n = 0; n < 4; ++n) {
      ABFrag bfr;
      const int d  = n * 16 + (lane & 15);
      const int ko = (lane >> 4) * 8;
      bfr.q[0] = *(const uint4*)&sV[d * LDV + ko];
      bfr.q[1] = *(const uint4*)&sV[d * LDV + ko + 16];
      acc[n] = wmma_bf16(pa, bfr, acc[n]);
    }
  }

  const int b = bn >> 4, h = bn & 15;
#pragma unroll
  for (int n = 0; n < 4; ++n) {
    const int d = n * 16 + (lane & 15);
#pragma unroll
    for (int r = 0; r < 8; ++r) {
      const float y = acc[n][r] / lrow[r];
      attnb[((size_t)b * SEQ + gr[r]) * CDIM + h * HD + d] = f2bf(y);
    }
  }
}

// ---------------- host launch ----------------
extern "C" void kernel_launch(void* const* d_in, const int* in_sizes, int n_in,
                              void* d_out, int out_size, void* d_ws, size_t ws_size,
                              hipStream_t stream) {
  const float* x  = (const float*)d_in[0];
  const float* Wq = (const float*)d_in[1];
  const float* bq = (const float*)d_in[2];
  const float* Wk = (const float*)d_in[3];
  const float* bk = (const float*)d_in[4];
  const float* Wv = (const float*)d_in[5];
  const float* bv = (const float*)d_in[6];
  const float* Wp = (const float*)d_in[7];
  const float* bp = (const float*)d_in[8];
  float* out = (float*)d_out;

  const size_t MK = (size_t)MROWS * CDIM;   // activations
  const size_t WK = (size_t)CDIM * CDIM;    // weight matrix

  unsigned short* xb  = (unsigned short*)d_ws;
  unsigned short* wqb = xb  + MK;
  unsigned short* wkb = wqb + WK;
  unsigned short* wvb = wkb + WK;
  unsigned short* wpb = wvb + WK;
  unsigned short* Qb  = wpb + WK;
  unsigned short* Kb  = Qb  + MK;
  unsigned short* Vb  = Kb  + MK;
  unsigned short* Ab  = Vb  + MK;           // total ~88 MB of d_ws

  cvt_kernel<<<2048, 256, 0, stream>>>(x,  xb,  (int)(MK / 4));
  cvt_kernel<<<1024, 256, 0, stream>>>(Wq, wqb, (int)(WK / 4));
  cvt_kernel<<<1024, 256, 0, stream>>>(Wk, wkb, (int)(WK / 4));
  cvt_kernel<<<1024, 256, 0, stream>>>(Wv, wvb, (int)(WK / 4));
  cvt_kernel<<<1024, 256, 0, stream>>>(Wp, wpb, (int)(WK / 4));

  dim3 gg(CDIM / 64, MROWS / 128);
  gemm_xwt<<<gg, 256, 0, stream>>>(xb, wqb, bq, Qb, nullptr, 0, 0.125f); // Q, 1/sqrt(64)
  gemm_xwt<<<gg, 256, 0, stream>>>(xb, wkb, bk, Kb, nullptr, 1, 1.0f);
  gemm_xwt<<<gg, 256, 0, stream>>>(xb, wvb, bv, Vb, nullptr, 2, 1.0f);

  attn_kernel<<<dim3(SEQ / 64, BATCH * NHEAD), 128, 0, stream>>>(Qb, Kb, Vb, Ab);

  gemm_xwt<<<gg, 256, 0, stream>>>(Ab, wpb, bp, nullptr, out, 3, 1.0f);
}
